// MultiCELoss_83760452206735
// MI455X (gfx1250) — matrix operand order repaired and checked by reading, not compiled, not run on hardware
//
#include <hip/hip_runtime.h>
#include <math.h>
#include <stdint.h>

#define B_ROWS   2048
#define R_PER    10
#define D_DIM    1024
#define N_TGT    (B_ROWS * (R_PER - 1))      // 18432 targets
#define NSPLIT   36
#define NCOLS    (N_TGT / NSPLIT)            // 512 columns per chunk
#define ALPHA_C  1.0f
#define BETA_C   0.5f
#define INV_TEMP 20.0f                       // 1 / 0.05

// LDS B panel: 64 cols x 1024 K bf16, with 4 bf16 (8B) pad every 512 bf16
// (TDM pad_interval=256 dwords, pad_amount=2 dwords) -> col stride 1032 bf16.
#define LDS_COL_STRIDE 1032

typedef __bf16 bf16_t;
typedef __attribute__((ext_vector_type(16))) __bf16 v16bf;
typedef __attribute__((ext_vector_type(8)))  float  v8f;
typedef __attribute__((ext_vector_type(4)))  unsigned int v4u;
typedef __attribute__((ext_vector_type(8)))  int v8i;
typedef __attribute__((ext_vector_type(4)))  int v4i;

union Frag { uint4 q[2]; v16bf v; };

// ---------------------------------------------------------------------------
// 1) Normalize each embedding row, convert to bf16, scatter into WMMA operand
//    buffers: a_bf[2048][1024] (anchors), t_bf[18432][1024] (pos then negs).
// ---------------------------------------------------------------------------
__global__ void norm_cvt_kernel(const float* __restrict__ emb,
                                bf16_t* __restrict__ a_bf,
                                bf16_t* __restrict__ t_bf) {
  __shared__ float red[256];
  const int row = blockIdx.x;          // 0 .. 20479
  const int tid = threadIdx.x;
  const float* src = emb + (size_t)row * D_DIM;
  float v[4];
  float ssq = 0.f;
#pragma unroll
  for (int c = 0; c < 4; ++c) { v[c] = src[tid + 256 * c]; ssq += v[c] * v[c]; }
  red[tid] = ssq; __syncthreads();
  for (int off = 128; off > 0; off >>= 1) {
    if (tid < off) red[tid] += red[tid + off];
    __syncthreads();
  }
  const float inv = 1.0f / fmaxf(sqrtf(red[0]), 1e-12f);
  const int b = row / R_PER, j = row % R_PER;
  bf16_t* dst;
  if (j == 0)      dst = a_bf + (size_t)b * D_DIM;                              // anchor
  else if (j == 1) dst = t_bf + (size_t)b * D_DIM;                              // pos
  else             dst = t_bf + (size_t)(B_ROWS + b * 8 + (j - 2)) * D_DIM;     // negs
#pragma unroll
  for (int c = 0; c < 4; ++c) dst[tid + 256 * c] = (bf16_t)(v[c] * inv);
}

// ---------------------------------------------------------------------------
// 2) Per-row KL term + exact f32 diagonal (pos cosine / TEMP). One block per b.
// ---------------------------------------------------------------------------
__global__ void rowscores_kernel(const float* __restrict__ emb,
                                 const float* __restrict__ scores,
                                 float* __restrict__ diag,
                                 float* __restrict__ kl_accum) {
  __shared__ float red[256];
  __shared__ float fin[19];
  const int b = blockIdx.x, tid = threadIdx.x;
  const float* a = emb + (size_t)(b * R_PER) * D_DIM;
  float ssq_a = 0.f, dot[9], ssq[9];
#pragma unroll
  for (int t = 0; t < 9; ++t) { dot[t] = 0.f; ssq[t] = 0.f; }
  for (int c = 0; c < 4; ++c) {
    const int d = tid + 256 * c;
    const float va = a[d];
    ssq_a += va * va;
#pragma unroll
    for (int t = 0; t < 9; ++t) {
      const float vt = a[(size_t)(t + 1) * D_DIM + d];
      dot[t] += va * vt;
      ssq[t] += vt * vt;
    }
  }
#pragma unroll
  for (int q = 0; q < 19; ++q) {
    const float val = (q == 0) ? ssq_a : (q < 10 ? dot[q - 1] : ssq[q - 10]);
    red[tid] = val; __syncthreads();
    for (int off = 128; off > 0; off >>= 1) {
      if (tid < off) red[tid] += red[tid + off];
      __syncthreads();
    }
    if (tid == 0) fin[q] = red[0];
    __syncthreads();
  }
  if (tid == 0) {
    const float inva = 1.f / fmaxf(sqrtf(fin[0]), 1e-12f);
    float sc[9], mx = -1e30f;
#pragma unroll
    for (int t = 0; t < 9; ++t) {
      sc[t] = fin[1 + t] * inva / fmaxf(sqrtf(fin[10 + t]), 1e-12f);
      mx = fmaxf(mx, sc[t]);
    }
    float se = 0.f;
#pragma unroll
    for (int t = 0; t < 9; ++t) se += __expf(sc[t] - mx);
    const float lse = mx + __logf(se);
    float s9[9], m2 = -1e30f;
#pragma unroll
    for (int t = 0; t < 9; ++t) { s9[t] = scores[b * 9 + t]; m2 = fmaxf(m2, s9[t]); }
    float se2 = 0.f;
#pragma unroll
    for (int t = 0; t < 9; ++t) { s9[t] = __expf(s9[t] - m2); se2 += s9[t]; }
    float kl = 0.f;
#pragma unroll
    for (int t = 0; t < 9; ++t) {
      const float ce = s9[t] / se2;
      kl += ce * __logf(ce) - ce * (sc[t] - lse);
    }
    atomicAdd(kl_accum, kl * (1.0f / 9.0f));
    diag[b] = sc[0] * INV_TEMP;   // sim[i,i] = cos(anchor, pos)/TEMP
  }
}

// ---------------------------------------------------------------------------
// 3) WMMA bf16 GEMM fused with online logsumexp.
//    Block = 4 waves covering 64 anchor rows; B panel (64 cols x 1024 K)
//    staged into LDS by one TDM tensor_load_to_lds per tile group and shared
//    by all 4 waves (x4 reduction in global B traffic). A comes straight from
//    the L2-resident anchor buffer. grid = (NSPLIT, B/64).
// ---------------------------------------------------------------------------
__global__ __launch_bounds__(128) void sim_lse_kernel(
    const bf16_t* __restrict__ a_bf,
    const bf16_t* __restrict__ t_bf,
    float* __restrict__ partial_lse) {
  // 64 cols * 1032 bf16 (incl. TDM pad of 4 bf16 per 512) = 132096 bytes.
  // Sole LDS object in this kernel -> LDS offset 0 (matches D#.lds_addr).
  __shared__ bf16_t ldsB[64 * LDS_COL_STRIDE];

  const int chunk = blockIdx.x;          // 0..35
  const int mblk  = blockIdx.y;          // 0..31 (64 rows each)
  const int tid   = threadIdx.x;
  const int wave  = tid >> 5;
  const int lane  = tid & 31;
  const int half  = lane >> 4;
  const int l15   = lane & 15;

  // A fragment (16-bit 16x32): lanes 0-15 hold K=[0..15], lanes 16-31
  // hold K=[16..31], for row M = lane&15.
  const bf16_t* aRow =
      a_bf + (size_t)(mblk * 64 + wave * 16 + l15) * D_DIM + half * 16;

  float m[8], s[8];
#pragma unroll
  for (int r = 0; r < 8; ++r) { m[r] = -1e30f; s[r] = 0.f; }

  for (int tg = 0; tg < NCOLS / 64; ++tg) {    // 8 tile groups of 64 columns
    const int col0 = chunk * NCOLS + tg * 64;
    __syncthreads();                           // previous panel fully consumed
    if (tid < 32) {
      // Build TDM descriptor: 2-D tile, 512 dwords (2048B) x 64 rows,
      // row-contiguous source, LDS pad 2 dwords per 256 dwords.
      const unsigned long long ga =
          (unsigned long long)(size_t)(t_bf + (size_t)col0 * D_DIM);
      v4u g0;
      g0[0] = 1u;                                       // count=1 (valid D#)
      g0[1] = 0u;                                       // lds_addr: ldsB @ 0
      g0[2] = (unsigned)ga;                             // global_addr[31:0]
      g0[3] = (unsigned)((ga >> 32) & 0x1FFFFFFull)     // global_addr[56:32]
              | (2u << 30);                             // type = 2 (image)
      v8i g1;
      g1[0] = (int)((2u << 16)      // data_size = 4B
                  | (1u << 20)      // pad_enable
                  | (7u << 22)      // pad_interval: every 256 dwords
                  | (1u << 25));    // pad_amount: 2 dwords
      g1[1] = (int)(512u << 16);    // tensor_dim0 = 512 dwords   (bits 79:48)
      g1[2] = (int)(64u  << 16);    // tensor_dim1 = 64 rows      (bits 111:80)
      g1[3] = (int)(512u << 16);    // tile_dim0   = 512 dwords   (bits 127:112)
      g1[4] = 64;                   // tile_dim1   = 64 rows      (bits 143:128)
      g1[5] = 512;                  // tensor_dim0_stride = 512 dwords
      g1[6] = 0;
      g1[7] = 0;
      v4i gz4 = {0, 0, 0, 0};
      v8i gz8 = {0, 0, 0, 0, 0, 0, 0, 0};
      __builtin_amdgcn_tensor_load_to_lds(g0, g1, gz4, gz4, gz8, 0);
      __builtin_amdgcn_s_wait_tensorcnt(0);
    }
    __syncthreads();                           // panel visible to all waves

    // The TDM wrote ldsB behind the compiler's back (the descriptor carries
    // the LDS address as an integer). Escape the pointer + memory clobber so
    // the ds_load fragment reads below cannot be folded to undef.
    asm volatile("" : : "r"(&ldsB[0]) : "memory");

    // B fragment (32x16): lane holds col N = lane&15, K = (lane>>4)*16 .. +15.
    const bf16_t* bp0 = ldsB + (size_t)( 0 + l15) * LDS_COL_STRIDE;
    const bf16_t* bp1 = ldsB + (size_t)(16 + l15) * LDS_COL_STRIDE;
    const bf16_t* bp2 = ldsB + (size_t)(32 + l15) * LDS_COL_STRIDE;
    const bf16_t* bp3 = ldsB + (size_t)(48 + l15) * LDS_COL_STRIDE;

    v8f acc0 = {}, acc1 = {}, acc2 = {}, acc3 = {};
#pragma unroll 4
    for (int k = 0; k < D_DIM; k += 32) {
      Frag A, B0, B1, B2, B3;
      A.q[0] = *(const uint4*)(aRow + k);
      A.q[1] = *(const uint4*)(aRow + k + 8);
      const int kk  = k + half * 16;
      const int off = kk + ((kk >> 9) << 2);   // +4 bf16 pad past K=512
      B0.q[0] = *(const uint4*)(bp0 + off);
      B0.q[1] = *(const uint4*)(bp0 + off + 8);
      B1.q[0] = *(const uint4*)(bp1 + off);
      B1.q[1] = *(const uint4*)(bp1 + off + 8);
      B2.q[0] = *(const uint4*)(bp2 + off);
      B2.q[1] = *(const uint4*)(bp2 + off + 8);
      B3.q[0] = *(const uint4*)(bp3 + off);
      B3.q[1] = *(const uint4*)(bp3 + off + 8);
      acc0 = __builtin_amdgcn_wmma_f32_16x16x32_bf16(false, A.v, false, B0.v, (short)0, acc0, false, false);
      acc1 = __builtin_amdgcn_wmma_f32_16x16x32_bf16(false, A.v, false, B1.v, (short)0, acc1, false, false);
      acc2 = __builtin_amdgcn_wmma_f32_16x16x32_bf16(false, A.v, false, B2.v, (short)0, acc2, false, false);
      acc3 = __builtin_amdgcn_wmma_f32_16x16x32_bf16(false, A.v, false, B3.v, (short)0, acc3, false, false);
    }

    // Online logsumexp update: lane holds C rows M = (lane>>4)*8 + r.
#pragma unroll
    for (int r = 0; r < 8; ++r) {
      {
        const float sim = acc0[r] * INV_TEMP;
        const float mn = fmaxf(m[r], sim);
        s[r] = s[r] * __expf(m[r] - mn) + __expf(sim - mn);
        m[r] = mn;
      }
      {
        const float sim = acc1[r] * INV_TEMP;
        const float mn = fmaxf(m[r], sim);
        s[r] = s[r] * __expf(m[r] - mn) + __expf(sim - mn);
        m[r] = mn;
      }
      {
        const float sim = acc2[r] * INV_TEMP;
        const float mn = fmaxf(m[r], sim);
        s[r] = s[r] * __expf(m[r] - mn) + __expf(sim - mn);
        m[r] = mn;
      }
      {
        const float sim = acc3[r] * INV_TEMP;
        const float mn = fmaxf(m[r], sim);
        s[r] = s[r] * __expf(m[r] - mn) + __expf(sim - mn);
        m[r] = mn;
      }
    }
  }

  // Butterfly combine across the 16 lanes of each half (tile columns).
#pragma unroll
  for (int off = 1; off < 16; off <<= 1) {
#pragma unroll
    for (int r = 0; r < 8; ++r) {
      const float mo = __shfl_xor(m[r], off, 32);
      const float so = __shfl_xor(s[r], off, 32);
      const float mn = fmaxf(m[r], mo);
      s[r] = s[r] * __expf(m[r] - mn) + so * __expf(mo - mn);
      m[r] = mn;
    }
  }
  if (l15 == 0) {
#pragma unroll
    for (int r = 0; r < 8; ++r) {
      const int row = mblk * 64 + wave * 16 + half * 8 + r;
      partial_lse[(size_t)row * NSPLIT + chunk] = m[r] + __logf(s[r]);
    }
  }
}

// ---------------------------------------------------------------------------
// 4) Combine chunk-LSEs per row, subtract diagonal, add KL term, emit loss.
// ---------------------------------------------------------------------------
__global__ void finalize_kernel(const float* __restrict__ partial_lse,
                                const float* __restrict__ diag,
                                const float* __restrict__ kl_accum,
                                float* __restrict__ out) {
  __shared__ float red[256];
  const int tid = threadIdx.x;
  float sum = 0.f;
  for (int row = tid; row < B_ROWS; row += 256) {
    const float* p = partial_lse + (size_t)row * NSPLIT;
    float mx = -1e30f;
    for (int c = 0; c < NSPLIT; ++c) mx = fmaxf(mx, p[c]);
    float se = 0.f;
    for (int c = 0; c < NSPLIT; ++c) se += __expf(p[c] - mx);
    sum += (mx + __logf(se)) - diag[row];
  }
  red[tid] = sum; __syncthreads();
  for (int off = 128; off > 0; off >>= 1) {
    if (tid < off) red[tid] += red[tid + off];
    __syncthreads();
  }
  if (tid == 0) {
    const float ce_mean = red[0] / (float)B_ROWS;
    const float kl_mean = kl_accum[0] / (float)B_ROWS;
    out[0] = BETA_C * kl_mean + ALPHA_C * ce_mean;
  }
}

__global__ void init_kernel(float* __restrict__ kl_accum) {
  if (threadIdx.x == 0) kl_accum[0] = 0.f;
}

extern "C" void kernel_launch(void* const* d_in, const int* in_sizes, int n_in,
                              void* d_out, int out_size, void* d_ws, size_t ws_size,
                              hipStream_t stream) {
  const float* emb    = (const float*)d_in[0];
  const float* scores = (const float*)d_in[1];
  (void)in_sizes; (void)n_in; (void)out_size; (void)ws_size;

  char* ws = (char*)d_ws;
  size_t off = 0;
  bf16_t* a_bf   = (bf16_t*)(ws + off); off += (size_t)B_ROWS * D_DIM * sizeof(bf16_t);
  bf16_t* t_bf   = (bf16_t*)(ws + off); off += (size_t)N_TGT  * D_DIM * sizeof(bf16_t);
  float* partial = (float*)(ws + off);  off += (size_t)B_ROWS * NSPLIT * sizeof(float);
  float* diag    = (float*)(ws + off);  off += (size_t)B_ROWS * sizeof(float);
  float* kl_acc  = (float*)(ws + off);  off += sizeof(float);

  init_kernel<<<1, 32, 0, stream>>>(kl_acc);
  norm_cvt_kernel<<<B_ROWS * R_PER, 256, 0, stream>>>(emb, a_bf, t_bf);
  rowscores_kernel<<<B_ROWS, 256, 0, stream>>>(emb, scores, diag, kl_acc);
  sim_lse_kernel<<<dim3(NSPLIT, B_ROWS / 64), 128, 0, stream>>>(a_bf, t_bf, partial);
  finalize_kernel<<<1, 256, 0, stream>>>(partial, diag, kl_acc, (float*)d_out);
}
